// Self_Attention_Adj_7224134992186
// MI455X (gfx1250) — matrix operand order repaired and checked by reading, compiled, not run on hardware
//
#include <hip/hip_runtime.h>
#include <hip/hip_bf16.h>

// Problem constants (from reference setup_inputs)
constexpr int kB = 8;     // batch
constexpr int kF = 128;   // features (GEMM-1 K dim)
constexpr int kN = 2048;  // tokens
constexpr int kA = 64;    // attention dim (GEMM-2 K dim)

typedef __attribute__((ext_vector_type(16))) __bf16 v16bf;
typedef __attribute__((ext_vector_type(8)))  __bf16 v8bf;
typedef __attribute__((ext_vector_type(8)))  float  v8f;

__device__ __forceinline__ float leaky(float v) { return v >= 0.f ? v : 0.01f * v; }

// ---------------------------------------------------------------------------
// Kernel 0: convert queue/key_w (f32 [F, A]) to bf16, pre-swizzled into the
// exact WMMA B-operand lane layout so the hot projection kernel fetches each
// B operand with a single 32-byte vector load.
// B-operand (32x16 bf16): lane half h holds rows 8h+i (i<8) and 16+8h+(i-8), col = lane%16.
// Storage: Bsw[((f0/32)*4 + j)*32 + lane][0..15]  (v16bf per lane).
// ---------------------------------------------------------------------------
__global__ __launch_bounds__(256) void pack_weights(const float* __restrict__ queue,
                                                    const float* __restrict__ key_w,
                                                    __bf16* __restrict__ Bq,
                                                    __bf16* __restrict__ Bk) {
  const int item = blockIdx.x * 256 + threadIdx.x;   // 0..511
  const int lane = item & 31;
  const int fj   = item >> 5;          // 0..15 : (f0-step, j)
  const int j    = fj & 3;
  const int f0   = (fj >> 2) * 32;
  const int m    = lane & 15;
  const int h    = lane >> 4;
  v16bf q, k;
#pragma unroll
  for (int i = 0; i < 16; ++i) {
    const int row = f0 + ((i < 8) ? (8 * h + i) : (16 + 8 * h + (i - 8)));
    const int col = j * 16 + m;
    q[i] = (__bf16)queue[row * kA + col];
    k[i] = (__bf16)key_w[row * kA + col];
  }
  *reinterpret_cast<v16bf*>(Bq + (size_t)item * 16) = q;
  *reinterpret_cast<v16bf*>(Bk + (size_t)item * 16) = k;
}

// ---------------------------------------------------------------------------
// Kernel 1: Q = leakyrelu(xt @ queue), K = leakyrelu(xt @ key_w), stored bf16.
// x is [B, F, N] so xt[b,n,f] = x[b*F*N + f*N + n].
// One wave per 16-row (n) tile; 8 waves per block.
// A-operand (16x32 bf16): lane half h holds K = 8h+i (i<8) and 16+8h+(i-8).
// x is read exactly once -> non-temporal loads (don't pollute L2).
// ---------------------------------------------------------------------------
__global__ __launch_bounds__(256) void proj_kernel(const float* __restrict__ x,
                                                   const __bf16* __restrict__ Bq,
                                                   const __bf16* __restrict__ Bk,
                                                   __bf16* __restrict__ Qw,
                                                   __bf16* __restrict__ Kw) {
  const int lane = threadIdx.x & 31;
  const int wave = threadIdx.x >> 5;
  const int tile = blockIdx.x * 8 + wave;      // 0..1023
  const int row0 = tile * 16;                  // flat row in [0, B*N)
  const int b    = row0 >> 11;                 // / kN
  const int n0   = row0 & (kN - 1);
  const int m    = lane & 15;
  const int h    = lane >> 4;

  const float* xb = x + (size_t)b * kF * kN;

  v8f qacc[4] = {};  // 4 column tiles of 16 (A = 64)
  v8f kacc[4] = {};

  for (int f0 = 0; f0 < kF; f0 += 32) {
    // Gather the transposed-x A operand (bf16 convert in flight)
    v16bf a;
#pragma unroll
    for (int i = 0; i < 16; ++i) {
      const int kk = (i < 8) ? (8 * h + i) : (16 + 8 * h + (i - 8));
      a[i] = (__bf16)__builtin_nontemporal_load(&xb[(size_t)(f0 + kk) * kN + n0 + m]);
    }
#pragma unroll
    for (int j = 0; j < 4; ++j) {
      const size_t bidx = (size_t)(((f0 >> 5) * 4 + j) * 32 + lane) * 16;
      const v16bf bq = *reinterpret_cast<const v16bf*>(Bq + bidx);
      const v16bf bk = *reinterpret_cast<const v16bf*>(Bk + bidx);
      qacc[j] = __builtin_amdgcn_wmma_f32_16x16x32_bf16(false, a, false, bq,
                                                        (short)0, qacc[j], false, false);
      kacc[j] = __builtin_amdgcn_wmma_f32_16x16x32_bf16(false, a, false, bk,
                                                        (short)0, kacc[j], false, false);
    }
  }

  // C/D layout: lane half h holds rows M = 8h+r, col = lane%16
#pragma unroll
  for (int j = 0; j < 4; ++j) {
#pragma unroll
    for (int r = 0; r < 8; ++r) {
      const int rowM  = 8 * h + r;
      const size_t id = (size_t)(row0 + rowM) * kA + j * 16 + m;
      Qw[id] = (__bf16)leaky(qacc[j][r]);
      Kw[id] = (__bf16)leaky(kacc[j][r]);
    }
  }
}

// ---------------------------------------------------------------------------
// Kernel 2: scores = Q @ K^T, softmax over rows (axis=1 of [B,N,N]), fused.
// Block = 256 thr = 8 waves = 2 column-strips of 16; 4 waves split the n-range
// (512 rows each), merge online (max,sum) via LDS, then re-GEMM + write
// exp(s - max) / sum.  Column m of a 16x16 C tile lives in lanes {m, m+16}.
// Output is streamed once -> non-temporal stores keep Q/K hot in L2.
// ---------------------------------------------------------------------------
__global__ __launch_bounds__(256) void attn_kernel(const __bf16* __restrict__ Qw,
                                                   const __bf16* __restrict__ Kw,
                                                   float* __restrict__ out) {
  __shared__ float lmax[2][4][16];
  __shared__ float lsum[2][4][16];

  const int lane = threadIdx.x & 31;
  const int wave = threadIdx.x >> 5;
  const int g    = wave >> 2;   // which column strip in block (0..1)
  const int s4   = wave & 3;    // n-quarter (0..3)
  const int m    = lane & 15;
  const int h    = lane >> 4;

  const int mtile = blockIdx.x * 2 + g;        // 0..1023
  const int b     = mtile >> 7;                // / (kN/16)
  const int m0    = (mtile & 127) << 4;

  const __bf16* Qb = Qw + (size_t)b * kN * kA;
  const __bf16* Kb = Kw + (size_t)b * kN * kA;

  // Preload B operand (K^T tile for columns m0..m0+15): 2 K-steps of 32.
  // Row of B = a index; lane half h holds rows st*32 + 16h .. +15 (contiguous).
  v16bf bop[2];
#pragma unroll
  for (int st = 0; st < 2; ++st)
    bop[st] = *reinterpret_cast<const v16bf*>(Kb + (size_t)(m0 + m) * kA + st * 32 + 16 * h);

  const int nlo = s4 * (kN / 4);
  const int nhi = nlo + (kN / 4);

  float rmax = -__builtin_inff();
  float rsum = 0.f;

  // ---- pass 1: online per-column (max, sum) over this wave's n-range ----
  for (int n0 = nlo; n0 < nhi; n0 += 16) {
    v8f acc = {};
    const __bf16* qrow = Qb + (size_t)(n0 + m) * kA;
#pragma unroll
    for (int st = 0; st < 2; ++st) {
      const v8bf lo = *reinterpret_cast<const v8bf*>(qrow + st * 32 + 16 * h);
      const v8bf hi = *reinterpret_cast<const v8bf*>(qrow + st * 32 + 16 * h + 8);
      v16bf a;
#pragma unroll
      for (int i = 0; i < 8; ++i) { a[i] = lo[i]; a[i + 8] = hi[i]; }
      acc = __builtin_amdgcn_wmma_f32_16x16x32_bf16(false, a, false, bop[st],
                                                    (short)0, acc, false, false);
    }
    float tmax = acc[0];
#pragma unroll
    for (int r = 1; r < 8; ++r) tmax = fmaxf(tmax, acc[r]);
    tmax = fmaxf(tmax, __shfl_xor(tmax, 16, 32));   // merge lane-halves (same column)
    const float nmax = fmaxf(rmax, tmax);
    float tsum = 0.f;
#pragma unroll
    for (int r = 0; r < 8; ++r) tsum += __expf(acc[r] - nmax);
    tsum += __shfl_xor(tsum, 16, 32);
    rsum = rsum * __expf(rmax - nmax) + tsum;
    rmax = nmax;
  }

  // ---- merge the 4 n-quarter partials through LDS ----
  if (lane < 16) { lmax[g][s4][m] = rmax; lsum[g][s4][m] = rsum; }
  __syncthreads();

  float fmx = lmax[g][0][m];
#pragma unroll
  for (int i = 1; i < 4; ++i) fmx = fmaxf(fmx, lmax[g][i][m]);
  float fsum = 0.f;
#pragma unroll
  for (int i = 0; i < 4; ++i) fsum += lsum[g][i][m] * __expf(lmax[g][i][m] - fmx);
  const float finv = 1.f / fsum;

  // ---- pass 2: re-GEMM and write normalized probabilities (NT stores) ----
  float* outb = out + (size_t)b * kN * kN;
  for (int n0 = nlo; n0 < nhi; n0 += 16) {
    v8f acc = {};
    const __bf16* qrow = Qb + (size_t)(n0 + m) * kA;
#pragma unroll
    for (int st = 0; st < 2; ++st) {
      const v8bf lo = *reinterpret_cast<const v8bf*>(qrow + st * 32 + 16 * h);
      const v8bf hi = *reinterpret_cast<const v8bf*>(qrow + st * 32 + 16 * h + 8);
      v16bf a;
#pragma unroll
      for (int i = 0; i < 8; ++i) { a[i] = lo[i]; a[i + 8] = hi[i]; }
      acc = __builtin_amdgcn_wmma_f32_16x16x32_bf16(false, a, false, bop[st],
                                                    (short)0, acc, false, false);
    }
#pragma unroll
    for (int r = 0; r < 8; ++r) {
      const int rowM = 8 * h + r;
      __builtin_nontemporal_store(__expf(acc[r] - fmx) * finv,
                                  &outb[(size_t)(n0 + rowM) * kN + m0 + m]);
    }
  }
}

// ---------------------------------------------------------------------------
extern "C" void kernel_launch(void* const* d_in, const int* in_sizes, int n_in,
                              void* d_out, int out_size, void* d_ws, size_t ws_size,
                              hipStream_t stream) {
  (void)in_sizes; (void)n_in; (void)out_size; (void)ws_size;
  const float* x     = (const float*)d_in[0];   // [B, F, N] f32
  const float* queue = (const float*)d_in[1];   // [F, A]    f32
  const float* key_w = (const float*)d_in[2];   // [F, A]    f32
  float* out = (float*)d_out;                   // [B, N, N] f32

  // Workspace: Q, K (bf16 [B*N, A], 2 MB each) + pre-swizzled weights (16 KB each)
  __bf16* Qw = (__bf16*)d_ws;
  __bf16* Kw = Qw + (size_t)kB * kN * kA;
  __bf16* Bq = Kw + (size_t)kB * kN * kA;
  __bf16* Bk = Bq + (size_t)kF * kA;

  pack_weights<<<2, 256, 0, stream>>>(queue, key_w, Bq, Bk);
  // 1024 row tiles / 8 waves per block
  proj_kernel<<<(kB * kN / 16) / 8, 256, 0, stream>>>(x, Bq, Bk, Qw, Kw);
  // 1024 column tiles / 2 per block
  attn_kernel<<<(kB * kN / 16) / 2, 256, 0, stream>>>(Qw, Kw, out);
}
